// New1_15771119911463
// MI455X (gfx1250) — compile-verified, matrix operand order checked by reading, not run on hardware
//
#include <hip/hip_runtime.h>
#include <hip/hip_fp16.h>

typedef __attribute__((ext_vector_type(16))) _Float16 v16h;
typedef __attribute__((ext_vector_type(8)))  float    v8f;

// ---------------------------------------------------------------------------
// Load one 16x32 f32->f16 fragment (CDNA5 ISA 7.12.2 layout) from a row-major
// matrix, caller guarantees in-bounds (K%32==0, row valid).
// ---------------------------------------------------------------------------
__device__ __forceinline__ v16h load_frag(const float* __restrict__ p)
{
    float4 a0 = *(const float4*)(p);
    float4 a1 = *(const float4*)(p + 4);
    float4 a2 = *(const float4*)(p + 16);
    float4 a3 = *(const float4*)(p + 20);
    v16h a;
    a[0]=(_Float16)a0.x; a[1]=(_Float16)a0.y; a[2]=(_Float16)a0.z; a[3]=(_Float16)a0.w;
    a[4]=(_Float16)a1.x; a[5]=(_Float16)a1.y; a[6]=(_Float16)a1.z; a[7]=(_Float16)a1.w;
    a[8]=(_Float16)a2.x; a[9]=(_Float16)a2.y; a[10]=(_Float16)a2.z; a[11]=(_Float16)a2.w;
    a[12]=(_Float16)a3.x; a[13]=(_Float16)a3.y; a[14]=(_Float16)a3.z; a[15]=(_Float16)a3.w;
    return a;
}

// ---------------------------------------------------------------------------
// Fused GEMM: out = [maybe relu]((A @ W^T + bias) * scale + shift) [+ res]
//   A: (M,K) f32 row-major, W: (>=gridDim.y*16, K) f32 row-major.
// Caller guarantees: M % 16 == 0, K % 32 == 0, W padded to the N-tile grid.
// One wave32 per 16x16 tile; branch-free K loop, 2 WMMAs per iteration.
// ---------------------------------------------------------------------------
__global__ __launch_bounds__(32)
void gemm_wmma_kernel(const float* __restrict__ A, const float* __restrict__ W,
                      const float* __restrict__ bias, const float* __restrict__ scale,
                      const float* __restrict__ shift, const float* __restrict__ res,
                      float* __restrict__ out, int M, int N, int K, int relu)
{
    const int lane = threadIdx.x & 31;
    const int half = lane >> 4;     // 0: lanes 0-15, 1: lanes 16-31
    const int l15  = lane & 15;
    const int row  = blockIdx.x * 16 + l15;   // A-fragment row (M)
    const int col  = blockIdx.y * 16 + l15;   // B-fragment col == W row (padded)

    const float* pa = A + (size_t)row * K + half * 8;
    const float* pb = W + (size_t)col * K + half * 8;

    v8f acc = {};
    int k0 = 0;
    for (; k0 + 64 <= K; k0 += 64) {
        v16h a0 = load_frag(pa);
        v16h b0 = load_frag(pb);
        v16h a1 = load_frag(pa + 32);
        v16h b1 = load_frag(pb + 32);
        acc = __builtin_amdgcn_wmma_f32_16x16x32_f16(false, a0, false, b0, (short)0, acc,
                                                     false, false);
        acc = __builtin_amdgcn_wmma_f32_16x16x32_f16(false, a1, false, b1, (short)0, acc,
                                                     false, false);
        pa += 64; pb += 64;
    }
    if (k0 < K) {  // K % 64 == 32 tail
        v16h a0 = load_frag(pa);
        v16h b0 = load_frag(pb);
        acc = __builtin_amdgcn_wmma_f32_16x16x32_f16(false, a0, false, b0, (short)0, acc,
                                                     false, false);
    }

    // Epilogue: C/D layout -> lane gives N=l15; VGPR r gives M = r + 8*half
    const int ocol = blockIdx.y * 16 + l15;
    if (ocol >= N) return;                    // only the padded final layer trips this
    const float sc = scale ? scale[ocol] : 1.0f;
    const float sh = shift ? shift[ocol] : 0.0f;
    const float bi = bias  ? bias[ocol]  : 0.0f;
    #pragma unroll
    for (int r = 0; r < 8; ++r) {
        const int orow = blockIdx.x * 16 + r + 8 * half;   // < M by construction
        float v = (acc[r] + bi) * sc + sh;
        if (relu) v = fmaxf(v, 0.0f);
        if (res) v += res[(size_t)orow * N + ocol];
        out[(size_t)orow * N + ocol] = v;
    }
}

// ---------------------------------------------------------------------------
// (B,3,N) -> coords (B*N,3) AND zero-padded embed input (B*N,32) so the
// embed GEMM runs with K=32 (no fallback path needed).
// ---------------------------------------------------------------------------
__global__ void transpose_x_kernel(const float* __restrict__ x, float* __restrict__ c,
                                   float* __restrict__ e0, int Nn, int total)
{
    int t = blockIdx.x * blockDim.x + threadIdx.x;
    if (t >= total) return;
    int b = t / Nn, n = t % Nn;
    float vx = x[((size_t)b * 3 + 0) * Nn + n];
    float vy = x[((size_t)b * 3 + 1) * Nn + n];
    float vz = x[((size_t)b * 3 + 2) * Nn + n];
    c[(size_t)t * 3 + 0] = vx;
    c[(size_t)t * 3 + 1] = vy;
    c[(size_t)t * 3 + 2] = vz;
    float* e = e0 + (size_t)t * 32;
    e[0] = vx; e[1] = vy; e[2] = vz;
    #pragma unroll
    for (int i = 3; i < 32; ++i) e[i] = 0.0f;
}

// ---------------------------------------------------------------------------
// Pad a (rows,K) weight to (rowsPad,K) with zero rows (for N=40 -> 48).
// ---------------------------------------------------------------------------
__global__ void pad_rows_kernel(const float* __restrict__ w, float* __restrict__ wp,
                                int rows, int K, int total)
{
    int t = blockIdx.x * blockDim.x + threadIdx.x;
    if (t >= total) return;
    int r = t / K;
    wp[t] = (r < rows) ? w[t] : 0.0f;
}

// ---------------------------------------------------------------------------
// Farthest point sampling: one workgroup per batch, min-dist array in LDS,
// argmax with first-occurrence tie-break (matches jnp.argmax).
// ---------------------------------------------------------------------------
__global__ __launch_bounds__(256)
void fps_kernel(const float* __restrict__ coords, int Np, int ns, int* __restrict__ idx)
{
    __shared__ float mind[1024];
    __shared__ float bv[256];
    __shared__ int   bi[256];
    __shared__ int   lastS;
    const int b = blockIdx.x, tid = threadIdx.x, nt = blockDim.x;
    for (int i = tid; i < Np; i += nt) mind[i] = 1e10f;
    if (tid == 0) lastS = 0;
    __syncthreads();
    for (int it = 0; it < ns; ++it) {
        const int last = lastS;
        if (tid == 0) idx[b * ns + it] = last;
        const float px = coords[((size_t)b * Np + last) * 3 + 0];
        const float py = coords[((size_t)b * Np + last) * 3 + 1];
        const float pz = coords[((size_t)b * Np + last) * 3 + 2];
        float bestv = -1.0f; int besti = 0x7fffffff;
        for (int i = tid; i < Np; i += nt) {
            float dx = coords[((size_t)b * Np + i) * 3 + 0] - px;
            float dy = coords[((size_t)b * Np + i) * 3 + 1] - py;
            float dz = coords[((size_t)b * Np + i) * 3 + 2] - pz;
            float d = dx * dx + dy * dy + dz * dz;
            float mv = fminf(mind[i], d);
            mind[i] = mv;
            if (mv > bestv || (mv == bestv && i < besti)) { bestv = mv; besti = i; }
        }
        bv[tid] = bestv; bi[tid] = besti;
        __syncthreads();
        for (int s = nt >> 1; s > 0; s >>= 1) {
            if (tid < s) {
                if (bv[tid + s] > bv[tid] ||
                    (bv[tid + s] == bv[tid] && bi[tid + s] < bi[tid])) {
                    bv[tid] = bv[tid + s]; bi[tid] = bi[tid + s];
                }
            }
            __syncthreads();
        }
        if (tid == 0) lastS = bi[0];
        __syncthreads();
    }
}

// ---------------------------------------------------------------------------
// Gather sampled rows. (k=0 neighbor of an FPS sample is the sample itself,
// and only k=0 survives the reference's local MLP, so kNN drops out.)
// ---------------------------------------------------------------------------
__global__ void gather_kernel(const float* __restrict__ coordsIn, const float* __restrict__ featsIn,
                              const int* __restrict__ idx, float* __restrict__ coordsOut,
                              float* __restrict__ featsOut, int NpIn, int P, int C, int total)
{
    int t = blockIdx.x * blockDim.x + threadIdx.x;
    if (t >= total) return;
    int b = t / P, p = t % P;
    int id = idx[b * P + p];
    coordsOut[(size_t)t * 3 + 0] = coordsIn[((size_t)b * NpIn + id) * 3 + 0];
    coordsOut[(size_t)t * 3 + 1] = coordsIn[((size_t)b * NpIn + id) * 3 + 1];
    coordsOut[(size_t)t * 3 + 2] = coordsIn[((size_t)b * NpIn + id) * 3 + 2];
    const float* src = featsIn + ((size_t)b * NpIn + id) * C;
    float* dst = featsOut + (size_t)t * C;
    for (int c = 0; c < C; ++c) dst[c] = src[c];
}

// ---------------------------------------------------------------------------
// Pack per-head weights (h, dh, d) into block-diagonal (h*dh, h*d) so Q/K/V
// become one big WMMA GEMM each.
// ---------------------------------------------------------------------------
__global__ void pack_blockdiag_kernel(const float* __restrict__ w, float* __restrict__ Wbig,
                                      int h, int dh, int d, int total)
{
    int t = blockIdx.x * blockDim.x + threadIdx.x;
    if (t >= total) return;
    int D = h * d;
    int rowi = t / D, c = t % D;
    int hi = rowi / dh, o = rowi % dh;
    float val = 0.0f;
    if (c >= hi * d && c < (hi + 1) * d)
        val = w[((size_t)hi * dh + o) * d + (c - hi * d)];
    Wbig[t] = val;
}

// ---------------------------------------------------------------------------
// Attention: one workgroup per (batch, head). K/V staged in LDS; each thread
// runs single-pass online softmax over its queries (exact flash recurrence).
// ---------------------------------------------------------------------------
template <int DH>
__global__ void attn_kernel(const float* __restrict__ q, const float* __restrict__ k,
                            const float* __restrict__ v, float* __restrict__ o,
                            int P, int heads)
{
    extern __shared__ float sm[];
    const int b = blockIdx.x, head = blockIdx.y;
    const int inner = heads * DH;
    float* Ks = sm;
    float* Vs = sm + (size_t)P * DH;
    const int tid = threadIdx.x, nt = blockDim.x;
    for (int i = tid; i < P * DH; i += nt) {
        int j = i / DH, d = i % DH;
        size_t base = ((size_t)(b * P + j)) * inner + head * DH + d;
        Ks[i] = k[base];
        Vs[i] = v[base];
    }
    __syncthreads();
    const float scale = rsqrtf((float)DH);
    for (int p = tid; p < P; p += nt) {
        float qr[DH], acc[DH];
        #pragma unroll
        for (int d = 0; d < DH; ++d) {
            qr[d] = q[((size_t)(b * P + p)) * inner + head * DH + d];
            acc[d] = 0.0f;
        }
        float m = -1e30f, l = 0.0f;
        for (int j = 0; j < P; ++j) {
            float s = 0.0f;
            #pragma unroll
            for (int d = 0; d < DH; ++d) s += qr[d] * Ks[j * DH + d];
            s *= scale;
            float mn = fmaxf(m, s);
            float corr = __expf(m - mn);
            float w = __expf(s - mn);
            l = l * corr + w;
            #pragma unroll
            for (int d = 0; d < DH; ++d) acc[d] = acc[d] * corr + w * Vs[j * DH + d];
            m = mn;
        }
        float inv = 1.0f / l;
        #pragma unroll
        for (int d = 0; d < DH; ++d)
            o[((size_t)(b * P + p)) * inner + head * DH + d] = acc[d] * inv;
    }
}

// ---------------------------------------------------------------------------
// Max-pool over points: out[b,c] = max_p f[b,p,c]
// ---------------------------------------------------------------------------
__global__ void maxpool_kernel(const float* __restrict__ f, float* __restrict__ out,
                               int P, int C, int total)
{
    int t = blockIdx.x * blockDim.x + threadIdx.x;
    if (t >= total) return;
    int b = t / C, c = t % C;
    float m = -1e30f;
    for (int p = 0; p < P; ++p) m = fmaxf(m, f[((size_t)b * P + p) * C + c]);
    out[t] = m;
}

// ---------------------------------------------------------------------------
extern "C" void kernel_launch(void* const* d_in, const int* in_sizes, int n_in,
                              void* d_out, int out_size, void* d_ws, size_t ws_size,
                              hipStream_t stream)
{
    (void)in_sizes; (void)n_in; (void)out_size;
    int pi = 0;
    auto NX = [&]() { return (const float*)d_in[pi++]; };

    const float* X = NX();
    const float *e_w1 = NX(), *e_b1 = NX(), *e_s1 = NX(), *e_t1 = NX();
    const float *e_w2 = NX(), *e_b2 = NX(), *e_s2 = NX(), *e_t2 = NX();

    const int cinA[4] = {32, 64, 128, 256};
    const int dimA[4] = {64, 128, 256, 512};
    const int hA[4]   = {8, 16, 16, 32};
    const int dhA[4]  = {16, 16, 32, 32};
    const int ptsA[4] = {512, 128, 32, 8};
    const int nbA[4]  = {1, 2, 1, 1};

    struct Blk { const float *wq, *wk, *wv, *wo, *so, *to, *wf, *bf, *sf, *tf; };
    struct Stg { const float *w1, *b1, *s1, *t1, *w2, *b2, *s2, *t2; Blk blk[2]; };
    Stg st[4];
    for (int s = 0; s < 4; ++s) {
        st[s].w1 = NX(); st[s].b1 = NX(); st[s].s1 = NX(); st[s].t1 = NX();
        st[s].w2 = NX(); st[s].b2 = NX(); st[s].s2 = NX(); st[s].t2 = NX();
        for (int j = 0; j < nbA[s]; ++j) {
            Blk& k = st[s].blk[j];
            k.wq = NX(); k.wk = NX(); k.wv = NX(); k.wo = NX(); k.so = NX();
            k.to = NX(); k.wf = NX(); k.bf = NX(); k.sf = NX(); k.tf = NX();
        }
    }
    const float *c_w1 = NX(), *c_b1 = NX(), *c_s1 = NX(), *c_t1 = NX();
    const float *c_w2 = NX(), *c_b2 = NX(), *c_s2 = NX(), *c_t2 = NX();
    const float *c_w3 = NX(), *c_b3 = NX();

    // ---- workspace layout (floats) ----
    float* ws = (float*)d_ws;
    size_t off = 0;
    auto alloc = [&](size_t n) { float* p = ws + off; off += n; return p; };
    float* coordsA = alloc(32 * 1024 * 3);
    float* coordsB = alloc(32 * 512 * 3);
    float* E0  = alloc(32 * 1024 * 32);   // zero-padded embed input (K=32)
    float* F   = alloc(32 * 1024 * 32);   // current features (max 1,048,576)
    float* G   = alloc(32 * 512 * 32);    // gathered feats / pooled
    float* H   = alloc(32 * 512 * 32);    // MLP temp
    float* Qb  = alloc(32 * 512 * 128);   // q (also embed temp / cls temp)
    float* Kb  = alloc(32 * 512 * 128);   // k
    float* Vb  = alloc(32 * 512 * 128);   // v
    float* Ob  = alloc(32 * 512 * 128);   // attention output
    float* ATT = alloc(32 * 512 * 64);    // post-wo residual stream
    float* Wq  = alloc(1024 * 512);       // packed block-diag weights
    float* Wk  = alloc(1024 * 512);
    float* Wv  = alloc(1024 * 512);
    float* W3p = alloc(48 * 256);         // padded final classifier weight
    int*   idxB = (int*)(ws + off); off += 32 * 512;
    if (ws_size < off * sizeof(float)) return;

    auto gemm = [&](const float* Ap, const float* Wp, const float* bi, const float* sc,
                    const float* sh, const float* rs, float* op, int M, int N, int K, int relu) {
        dim3 g((M + 15) / 16, (N + 15) / 16);
        gemm_wmma_kernel<<<g, dim3(32), 0, stream>>>(Ap, Wp, bi, sc, sh, rs, op, M, N, K, relu);
    };

    // ---- embed: (B*N,32 padded) -> 32 -> 32 ----
    {
        int tot = 32 * 1024;
        transpose_x_kernel<<<(tot + 255) / 256, 256, 0, stream>>>(X, coordsA, E0, 1024, tot);
    }
    gemm(E0, e_w1, e_b1, e_s1, e_t1, nullptr, Qb, 32768, 32, 32, 1);
    gemm(Qb, e_w2, e_b2, e_s2, e_t2, nullptr, F, 32768, 32, 32, 1);

    const float* ccur = coordsA;
    float* cnext = coordsB;
    int Np = 1024, Ccur = 32;

    for (int s = 0; s < 4; ++s) {
        const int Pp = ptsA[s], cin = cinA[s], dim = dimA[s];
        const int hh = hA[s], dh = dhA[s], inner = hh * dh, M = 32 * Pp;

        fps_kernel<<<32, 256, 0, stream>>>(ccur, Np, Pp, idxB);
        {
            int tot = 32 * Pp;
            gather_kernel<<<(tot + 127) / 128, 128, 0, stream>>>(ccur, F, idxB, cnext, G,
                                                                 Np, Pp, Ccur, tot);
        }
        // local MLP: cin -> cin -> dim (only k=0 element survives in reference)
        gemm(G, st[s].w1, st[s].b1, st[s].s1, st[s].t1, nullptr, H, M, cin, cin, 1);
        gemm(H, st[s].w2, st[s].b2, st[s].s2, st[s].t2, nullptr, F, M, dim, cin, 1);

        for (int j = 0; j < nbA[s]; ++j) {
            const Blk& bk = st[s].blk[j];
            const int packN = inner * dim, d = dim / hh;
            pack_blockdiag_kernel<<<(packN + 255) / 256, 256, 0, stream>>>(bk.wq, Wq, hh, dh, d, packN);
            pack_blockdiag_kernel<<<(packN + 255) / 256, 256, 0, stream>>>(bk.wk, Wk, hh, dh, d, packN);
            pack_blockdiag_kernel<<<(packN + 255) / 256, 256, 0, stream>>>(bk.wv, Wv, hh, dh, d, packN);
            gemm(F, Wq, nullptr, nullptr, nullptr, nullptr, Qb, M, inner, dim, 0);
            gemm(F, Wk, nullptr, nullptr, nullptr, nullptr, Kb, M, inner, dim, 0);
            gemm(F, Wv, nullptr, nullptr, nullptr, nullptr, Vb, M, inner, dim, 0);
            size_t shb = (size_t)2 * Pp * dh * sizeof(float);
            if (dh == 16)
                attn_kernel<16><<<dim3(32, hh), 128, shb, stream>>>(Qb, Kb, Vb, Ob, Pp, hh);
            else
                attn_kernel<32><<<dim3(32, hh), 128, shb, stream>>>(Qb, Kb, Vb, Ob, Pp, hh);
            // att = relu(bn(o @ wo^T)) + f
            gemm(Ob, bk.wo, nullptr, bk.so, bk.to, F, ATT, M, dim, inner, 1);
            // f' = relu(bn(att @ wf^T + bf)) + att
            gemm(ATT, bk.wf, bk.bf, bk.sf, bk.tf, ATT, F, M, dim, dim, 1);
        }
        const float* t = ccur; ccur = cnext; cnext = (float*)t;
        Np = Pp; Ccur = dim;
    }

    // ---- global max pool over the 8 remaining points, then classifier ----
    {
        int tot = 32 * 512;
        maxpool_kernel<<<(tot + 255) / 256, 256, 0, stream>>>(F, G, 8, 512, tot);
    }
    gemm(G, c_w1, c_b1, c_s1, c_t1, nullptr, H, 32, 256, 512, 1);
    gemm(H, c_w2, c_b2, c_s2, c_t2, nullptr, Qb, 32, 256, 256, 1);
    {
        int tot = 48 * 256;
        pad_rows_kernel<<<(tot + 255) / 256, 256, 0, stream>>>(c_w3, W3p, 40, 256, tot);
    }
    gemm(Qb, W3p, c_b3, nullptr, nullptr, nullptr, (float*)d_out, 32, 40, 256, 0);
}
